// Head_2654289789032
// MI455X (gfx1250) — matrix operand order repaired and checked by reading, not compiled, not run on hardware
//
#include <hip/hip_runtime.h>
#include <hip/hip_bf16.h>

// Single-head causal attention for MI455X (gfx1250, wave32, WMMA).
// Pipeline: (1) f16 QKV projection GEMM via v_wmma_f32_16x16x32_f16,
//           (2) attention: S=QK^T (WMMA) -> softmax (f32) -> O=P.V (WMMA).

typedef __attribute__((ext_vector_type(16))) _Float16 v16h;
typedef __attribute__((ext_vector_type(8)))  _Float16 v8h;
typedef __attribute__((ext_vector_type(8)))  float    v8f;

#define NB   256      // batch
#define NT   256      // sequence length
#define NC   384      // embedding dim
#define NH   384      // head size

__device__ __forceinline__ v8f wmma_f16(v16h a, v16h b, v8f c) {
  // (neg_a, A, neg_b, B, c_mod, C, reuse_a, reuse_b)
  return __builtin_amdgcn_wmma_f32_16x16x32_f16(false, a, false, b,
                                                (short)0, c, false, false);
}

// A-fragment (16x32 f16, M x K): lane half 0 holds K {0..15}, half 1 {16..31}.
__device__ __forceinline__ v16h frag_a(const _Float16* p0, const _Float16* p1) {
  v16h r;
#pragma unroll
  for (int i = 0; i < 8; ++i) { r[i] = p0[i]; r[i + 8] = p1[i]; }
  return r;
}

// B-fragment (32x16 f16, K x N): lane = N, 16 contiguous K values.
__device__ __forceinline__ v16h frag_b(const _Float16* p) {
  v16h r;
#pragma unroll
  for (int i = 0; i < 16; ++i) r[i] = p[i];
  return r;
}

// ---------------------------------------------------------------------------
// Kernel 1: {K,Q,V} = X @ {Wk,Wq,Wv}^T  (M=B*T=65536, N=384, K=384).
// One workgroup owns a 32-row X tile: stage X->LDS(f16) ONCE, then loop all
// 3 weights x 6 N-tiles, restaging only the L2-hot W K-chunks. X is read
// from HBM exactly once (100 MB) instead of 18x.
// Block: 256 threads = 8 waves as 2 (M) x 4 (N) sub-tiles of 16x16.
// ---------------------------------------------------------------------------
#define K1_KC   128
#define K1_SX   392        // X LDS stride (halves): 784B rows, 16B multiple
#define K1_SW   136        // W LDS stride (halves): 272B rows
__global__ void __launch_bounds__(256)
qkv_proj_kernel(const float* __restrict__ x,
                const float* __restrict__ Wk,
                const float* __restrict__ Wq,
                const float* __restrict__ Wv,
                _Float16* __restrict__ Kh,
                _Float16* __restrict__ Qh,
                _Float16* __restrict__ Vh) {
  __shared__ _Float16 xs[32 * K1_SX];   // 25088 B (full K=384)
  __shared__ _Float16 wsm[64 * K1_SW];  // 17408 B (one 128-K chunk)

  const int tid  = threadIdx.x;
  const int wave = tid >> 5;
  const int lane = tid & 31;
  const int m    = lane & 15;
  const int hf   = lane >> 4;
  const int wm   = wave >> 2;           // 0..1 (M sub-tile)
  const int wn   = wave & 3;            // 0..3 (N sub-tile)
  const int m0   = blockIdx.x * 32;

  // Stage full X tile once: 32 rows x 384 cols, f32 -> f16 (3072 float4)
#pragma unroll
  for (int j = 0; j < 12; ++j) {
    int li  = j * 256 + tid;
    int row = li / 96;                  // 96 float4 per row
    int c4  = (li % 96) * 4;
    float4 v = *(const float4*)(x + (size_t)(m0 + row) * NC + c4);
    xs[row * K1_SX + c4 + 0] = (_Float16)v.x;
    xs[row * K1_SX + c4 + 1] = (_Float16)v.y;
    xs[row * K1_SX + c4 + 2] = (_Float16)v.z;
    xs[row * K1_SX + c4 + 3] = (_Float16)v.w;
  }

#pragma unroll 1
  for (int z = 0; z < 3; ++z) {
    const float* __restrict__ W = (z == 0) ? Wk : ((z == 1) ? Wq : Wv);
    _Float16* __restrict__ Y    = (z == 0) ? Kh : ((z == 1) ? Qh : Vh);
    const float scale = (z == 1) ? 0.05103103630798288f : 1.0f; // 1/sqrt(384)

#pragma unroll 1
    for (int n0 = 0; n0 < NH; n0 += 64) {
      v8f acc = {};
#pragma unroll
      for (int kb = 0; kb < NC; kb += K1_KC) {
        __syncthreads();                // previous chunk's readers done
        // Stage W chunk: 64 rows x 128 cols (row n = (W^T) column n)
#pragma unroll
        for (int j = 0; j < 8; ++j) {
          int li  = j * 256 + tid;      // 2048 float4 total
          int row = li >> 5;
          int c4  = (li & 31) * 4;
          float4 v = *(const float4*)(W + (size_t)(n0 + row) * NC + kb + c4);
          wsm[row * K1_SW + c4 + 0] = (_Float16)v.x;
          wsm[row * K1_SW + c4 + 1] = (_Float16)v.y;
          wsm[row * K1_SW + c4 + 2] = (_Float16)v.z;
          wsm[row * K1_SW + c4 + 3] = (_Float16)v.w;
        }
        __syncthreads();
#pragma unroll
        for (int kc = 0; kc < K1_KC; kc += 32) {
          const _Float16* pa = &xs[(wm * 16 + m) * K1_SX + kb + kc + hf * 16];
          v16h a = frag_a(pa, pa + 8);
          v16h b = frag_b(&wsm[(wn * 16 + m) * K1_SW + kc + hf * 16]);
          acc = wmma_f16(a, b, acc);
        }
      }
      // C/D layout: VGPR r -> M = r + 8*hf, N = lane&15
#pragma unroll
      for (int r = 0; r < 8; ++r) {
        int row = m0 + wm * 16 + r + 8 * hf;
        int col = n0 + wn * 16 + m;
        Y[(size_t)row * NH + col] = (_Float16)(acc[r] * scale);
      }
    }
  }
}

// ---------------------------------------------------------------------------
// Kernel 2: causal attention on f16 Q/K/V, f32 output.
// Grid: (T/16, B); block 256 (8 waves). Each block: 16 query rows.
//   waves compute S=QK^T tiles (skip masked key tiles), softmax (f32),
//   then O = P.V with V^T staged per 32-key chunk. H split 8 ways (48/wave).
// ---------------------------------------------------------------------------
#define SQ 392   // lds_q stride (halves)
#define SS 260   // lds_s stride (f32)
#define SP 264   // lds_p stride (halves)
#define SV 40    // lds_vt stride (halves), rows = H columns

__global__ void __launch_bounds__(256)
attn_kernel(const _Float16* __restrict__ Qh,
            const _Float16* __restrict__ Kh,
            const _Float16* __restrict__ Vh,
            float* __restrict__ out) {
  // Phase-disjoint LDS: [q 12544B][p 8448B][s(f32,16640B) aliased with vt(30720B)]
  __shared__ alignas(16) unsigned char smem[12544 + 8448 + 30720];
  _Float16* lds_q  = (_Float16*)smem;
  _Float16* lds_p  = (_Float16*)(smem + 12544);
  float*    lds_s  = (float*)   (smem + 12544 + 8448);
  _Float16* lds_vt = (_Float16*)(smem + 12544 + 8448);

  const int tid   = threadIdx.x;
  const int wave  = tid >> 5;
  const int lane  = tid & 31;
  const int m     = lane & 15;
  const int hf    = lane >> 4;
  const int q_blk = blockIdx.x;          // 0..15
  const int b     = blockIdx.y;          // 0..255
  const int q0    = q_blk * 16;
  const int bT    = b * NT;

  // ---- Phase 1: stage Q tile (16 x 384 halves) ----
#pragma unroll
  for (int j = 0; j < 3; ++j) {
    int li  = (j * 256 + tid) * 8;       // 6144 halves total
    int row = li / NH;
    int hc  = li % NH;
    v8h v = *(const v8h*)(Qh + (size_t)(bT + q0 + row) * NH + hc);
#pragma unroll
    for (int e = 0; e < 8; ++e) lds_q[row * SQ + hc + e] = v[e];
  }
  __syncthreads();

  // ---- Phase 2: S = Q K^T (Q pre-scaled). Wave-uniform causal skip. ----
#pragma unroll
  for (int tt = 0; tt < 2; ++tt) {
    int t = wave + tt * 8;               // key tile 0..15
    if (t <= q_blk) {
      int key0 = t * 16;
      v8f acc = {};
#pragma unroll
      for (int kc = 0; kc < NH; kc += 32) {
        const _Float16* pa = &lds_q[m * SQ + kc + hf * 16];
        v16h a = frag_a(pa, pa + 8);
        // B[k][n] = K[key0+n][kc+k]: 16 contiguous halves per lane (hits L2)
        v16h bb = frag_b(Kh + (size_t)(bT + key0 + m) * NH + kc + hf * 16);
        acc = wmma_f16(a, bb, acc);
      }
#pragma unroll
      for (int r = 0; r < 8; ++r)
        lds_s[(r + 8 * hf) * SS + key0 + m] = acc[r];
    }
  }
  __syncthreads();

  // ---- Phase 3: softmax, wave handles rows 2w, 2w+1 (32 lanes x 8 keys) ----
#pragma unroll
  for (int rr = 0; rr < 2; ++rr) {
    int r    = wave * 2 + rr;
    int qrow = q0 + r;
    float sv[8];
    float mx = -3.0e38f;
#pragma unroll
    for (int j = 0; j < 8; ++j) {
      int k = j * 32 + lane;
      float s = (k <= qrow) ? lds_s[r * SS + k] : -3.0e38f;
      sv[j] = s;
      mx = fmaxf(mx, s);
    }
#pragma unroll
    for (int off = 16; off > 0; off >>= 1)
      mx = fmaxf(mx, __shfl_xor(mx, off, 32));
    float ex[8];
    float l = 0.0f;
#pragma unroll
    for (int j = 0; j < 8; ++j) {
      float e = __expf(sv[j] - mx);      // masked lanes underflow to 0
      ex[j] = e;
      l += e;
    }
#pragma unroll
    for (int off = 16; off > 0; off >>= 1)
      l += __shfl_xor(l, off, 32);
    float inv = 1.0f / l;                // l >= 1 (diagonal term)
#pragma unroll
    for (int j = 0; j < 8; ++j)
      lds_p[r * SP + j * 32 + lane] = (_Float16)(ex[j] * inv);
  }
  __syncthreads();                       // p done; s region now reusable as vt

  // ---- Phase 4: O = P . V, chunked over keys (32/chunk), H split by wave ----
  const int nch = (q_blk >> 1) + 1;      // chunks covering keys <= q0+15
  const int h0  = wave * 48;
  v8f o0 = {}, o1 = {}, o2 = {};
  for (int c = 0; c < nch; ++c) {
    int kc2 = c * 32;
    // Stage V^T chunk: lds_vt[h][key_local], 32 keys x 384 cols
#pragma unroll
    for (int j = 0; j < 6; ++j) {
      int li  = (j * 256 + tid) * 8;     // 12288 halves total
      int key = li / NH;
      int hc  = li % NH;
      v8h v = *(const v8h*)(Vh + (size_t)(bT + kc2 + key) * NH + hc);
#pragma unroll
      for (int e = 0; e < 8; ++e) lds_vt[(hc + e) * SV + key] = v[e];
    }
    if (c + 1 < nch)                     // gfx1250 global_prefetch_b8
      __builtin_prefetch(Vh + (size_t)(bT + kc2 + 32) * NH + (tid & 63) * 128, 0, 0);
    __syncthreads();
    const _Float16* pa = &lds_p[m * SP + kc2 + hf * 16];
    v16h a = frag_a(pa, pa + 8);
    o0 = wmma_f16(a, frag_b(&lds_vt[(h0 +  0 + m) * SV + hf * 16]), o0);
    o1 = wmma_f16(a, frag_b(&lds_vt[(h0 + 16 + m) * SV + hf * 16]), o1);
    o2 = wmma_f16(a, frag_b(&lds_vt[(h0 + 32 + m) * SV + hf * 16]), o2);
    __syncthreads();
  }

  // ---- Phase 5: store O (f32) ----
#pragma unroll
  for (int r = 0; r < 8; ++r) {
    size_t row = (size_t)(bT + q0 + r + 8 * hf);
    out[row * NH + h0 +  0 + m] = o0[r];
    out[row * NH + h0 + 16 + m] = o1[r];
    out[row * NH + h0 + 32 + m] = o2[r];
  }
}

extern "C" void kernel_launch(void* const* d_in, const int* in_sizes, int n_in,
                              void* d_out, int out_size, void* d_ws, size_t ws_size,
                              hipStream_t stream) {
  (void)in_sizes; (void)n_in; (void)out_size; (void)ws_size;
  const float* x  = (const float*)d_in[0];
  const float* Wk = (const float*)d_in[1];
  const float* Wq = (const float*)d_in[2];
  const float* Wv = (const float*)d_in[3];
  float* out = (float*)d_out;

  const size_t elems = (size_t)NB * NT * NH;          // 25,165,824
  _Float16* Kh = (_Float16*)d_ws;                     // 48 MB each, f16
  _Float16* Qh = Kh + elems;
  _Float16* Vh = Qh + elems;

  // 1) QKV projection (Q pre-scaled by 1/sqrt(H)); X read from HBM once.
  dim3 g1((NB * NT) / 32);
  qkv_proj_kernel<<<g1, 256, 0, stream>>>(x, Wk, Wq, Wv, Kh, Qh, Vh);

  // 2) Causal attention
  dim3 g2(NT / 16, NB);
  attn_kernel<<<g2, 256, 0, stream>>>(Qh, Kh, Vh, out);
}
